// RCNNNet_58076547776802
// MI455X (gfx1250) — compile-verified, matrix operand order checked by reading, not run on hardware
//
#include <hip/hip_runtime.h>
#include <hip/hip_bf16.h>

typedef _Float16 half8 __attribute__((ext_vector_type(8)));
typedef _Float16 v16h  __attribute__((ext_vector_type(16)));
typedef float    v8f   __attribute__((ext_vector_type(8)));

__device__ __forceinline__ v8f v8f_zero() {
    v8f z = {0.f,0.f,0.f,0.f,0.f,0.f,0.f,0.f};
    return z;
}

// Load one 16x32 f16 tile (A- or B-operand layout; identical lane pattern).
// p points at tile[0][0]; ld = row stride in halfs (multiple of 8).
__device__ __forceinline__ v16h ld_tile(const _Float16* p, int ld) {
    const int lane = threadIdx.x & 31;
    const int r  = lane & 15;
    const int hi = lane >> 4;
    const half8* q0 = reinterpret_cast<const half8*>(p + (size_t)r * ld + hi * 8);
    const half8* q1 = reinterpret_cast<const half8*>(p + (size_t)r * ld + 16 + hi * 8);
    union { v16h v; half8 h[2]; } u;
    u.h[0] = *q0;
    u.h[1] = *q1;
    return u.v;
}

// acc[MT] += A(16*MT x 32*KT, LDS) x W(ntile, K)^T  via v_wmma_f32_16x16x32_f16
template <int MT>
__device__ __forceinline__ void wmma_block(const _Float16* A, int lda,
                                           const _Float16* W, int Kpad,
                                           int ntile, int KT, v8f* acc) {
    for (int kt = 0; kt < KT; ++kt) {
        v16h bt = ld_tile(W + (size_t)(ntile * 16) * Kpad + kt * 32, Kpad);
#pragma unroll
        for (int mt = 0; mt < MT; ++mt) {
            v16h at = ld_tile(A + (size_t)(mt * 16) * lda + kt * 32, lda);
            acc[mt] = __builtin_amdgcn_wmma_f32_16x16x32_f16(
                false, at, false, bt, (short)0, acc[mt], false, false);
        }
    }
}

// bias + relu + scatter C tiles into an LDS activation buffer
__device__ __forceinline__ void store_relu_lds(_Float16* Ad, int ldd, int nt,
                                               const float* bias, const v8f* acc, int MT) {
    const int lane = threadIdx.x & 31;
    const int col  = lane & 15;
    const int hi   = lane >> 4;
    const float bb = bias[nt * 16 + col];
    for (int mt = 0; mt < MT; ++mt) {
#pragma unroll
        for (int v = 0; v < 8; ++v) {
            float x = acc[mt][v] + bb;
            x = x > 0.f ? x : 0.f;
            Ad[(size_t)(mt * 16 + v + 8 * hi) * ldd + nt * 16 + col] = (_Float16)x;
        }
    }
}

// ---------------- weight convert: f32 [N][K] -> f16 [N][Kpad] (zero pad) ----
__global__ void k_convpad(const float* __restrict__ src, _Float16* __restrict__ dst,
                          int N, int K, int Kpad) {
    int i = blockIdx.x * blockDim.x + threadIdx.x;
    int total = N * Kpad;
    if (i >= total) return;
    int n = i / Kpad, k = i % Kpad;
    dst[i] = (_Float16)((k < K) ? src[(size_t)n * K + k] : 0.f);
}

// ---------------- stage 1: per-point MLP 5->128->128, merge 256->128 --------
__global__ __launch_bounds__(256) void k_pointmlp(const float* __restrict__ pts,
        const _Float16* W0, const float* b0,
        const _Float16* W1, const float* b1,
        const _Float16* Wm, const float* bm,
        _Float16* __restrict__ feat0) {
    __shared__ __align__(16) _Float16 A0[64 * 40];
    __shared__ __align__(16) _Float16 A1[64 * 136];
    __shared__ __align__(16) _Float16 A2[64 * 264];
    const int tid = threadIdx.x;
    const int r0  = blockIdx.x * 64;

    for (int p = tid; p < 64 * 32; p += 256) {
        int row = p >> 5, k = p & 31;
        A0[row * 40 + k] = (_Float16)((k < 5) ? pts[(size_t)(r0 + row) * 133 + k] : 0.f);
    }
    for (int p = tid; p < 64 * 128; p += 256) {
        int row = p >> 7, k = p & 127;
        A2[row * 264 + 128 + k] = (_Float16)pts[(size_t)(r0 + row) * 133 + 5 + k];
    }
    __syncthreads();
    const int w = tid >> 5;
    {
        v8f acc[4] = {v8f_zero(), v8f_zero(), v8f_zero(), v8f_zero()};
        wmma_block<4>(A0, 40, W0, 32, w, 1, acc);
        store_relu_lds(A1, 136, w, b0, acc, 4);
    }
    __syncthreads();
    {
        v8f acc[4] = {v8f_zero(), v8f_zero(), v8f_zero(), v8f_zero()};
        wmma_block<4>(A1, 136, W1, 128, w, 4, acc);
        store_relu_lds(A2, 264, w, b1, acc, 4);
    }
    __syncthreads();
    {
        v8f acc[4] = {v8f_zero(), v8f_zero(), v8f_zero(), v8f_zero()};
        wmma_block<4>(A2, 264, Wm, 256, w, 8, acc);
        const int lane = tid & 31;
        const int col  = w * 16 + (lane & 15);
        const int hi   = lane >> 4;
        const float bb = bm[col];
        for (int mt = 0; mt < 4; ++mt) {
#pragma unroll
            for (int v = 0; v < 8; ++v) {
                float x = acc[mt][v] + bb;
                x = x > 0.f ? x : 0.f;
                feat0[(size_t)(r0 + mt * 16 + v + 8 * hi) * 128 + col] = (_Float16)x;
            }
        }
    }
}

// ---------------- farthest point sampling (one block per batch) -------------
__global__ __launch_bounds__(256) void k_fps(const float* __restrict__ xyz, int xs, int xbs,
                                             int N, int S, int* __restrict__ out) {
    __shared__ float dist[512];
    __shared__ float rv[256];
    __shared__ int   ri[256];
    __shared__ int   sfar;
    const int tid = threadIdx.x;
    const int b   = blockIdx.x;
    const float* base = xyz + (size_t)b * xbs;
    for (int i = tid; i < N; i += 256) dist[i] = 1e10f;
    __syncthreads();
    int far = 0;
    for (int s = 0; s < S; ++s) {
        if (tid == 0) out[(size_t)b * S + s] = far;
        const float cx = base[(size_t)far * xs + 0];
        const float cy = base[(size_t)far * xs + 1];
        const float cz = base[(size_t)far * xs + 2];
        float bv = -1.f; int bi = 0;
        for (int i = tid; i < N; i += 256) {
            float dx = base[(size_t)i * xs + 0] - cx;
            float dy = base[(size_t)i * xs + 1] - cy;
            float dz = base[(size_t)i * xs + 2] - cz;
            float d  = dx * dx + dy * dy + dz * dz;
            float dd = fminf(dist[i], d);
            dist[i] = dd;
            if (dd > bv || (dd == bv && i < bi)) { bv = dd; bi = i; }
        }
        rv[tid] = bv; ri[tid] = bi;
        __syncthreads();
        for (int st = 128; st > 0; st >>= 1) {
            if (tid < st) {
                float ov = rv[tid + st]; int oi = ri[tid + st];
                if (ov > rv[tid] || (ov == rv[tid] && oi < ri[tid])) { rv[tid] = ov; ri[tid] = oi; }
            }
            __syncthreads();
        }
        if (tid == 0) sfar = ri[0];
        __syncthreads();
        far = sfar;
        __syncthreads();
    }
}

__global__ void k_gather_nx(const float* __restrict__ xyz, int xs, int xbs,
                            const int* __restrict__ idx, int S,
                            float* __restrict__ nxyz, int total) {
    int i = blockIdx.x * blockDim.x + threadIdx.x;
    if (i >= total) return;
    int b = i / S;
    int id = idx[i];
    for (int k = 0; k < 3; ++k)
        nxyz[(size_t)i * 3 + k] = xyz[(size_t)b * xbs + (size_t)id * xs + k];
}

// ---------------- ball query (one thread per group, ascending scan) ---------
__global__ void k_ballq(const float* __restrict__ xyz, int xs, int xbs, int N,
                        const float* __restrict__ nxyz, int S, float r2,
                        int* __restrict__ out, int total) {
    int i = blockIdx.x * blockDim.x + threadIdx.x;
    if (i >= total) return;
    int b = i / S;
    const float* base = xyz + (size_t)b * xbs;
    float nx = nxyz[(size_t)i * 3 + 0];
    float ny = nxyz[(size_t)i * 3 + 1];
    float nz = nxyz[(size_t)i * 3 + 2];
    int cnt = 0, first = N - 1;
    int* o = out + (size_t)i * 64;
    for (int j = 0; j < N && cnt < 64; ++j) {
        float dx = base[(size_t)j * xs + 0] - nx;
        float dy = base[(size_t)j * xs + 1] - ny;
        float dz = base[(size_t)j * xs + 2] - nz;
        if (dx * dx + dy * dy + dz * dz < r2) {
            if (cnt == 0) first = j;
            o[cnt++] = j;
        }
    }
    for (int k = cnt; k < 64; ++k) o[k] = first;
}

// ---------------- fused SA module: gather -> MLP(131->128->128->C3) -> max --
__global__ __launch_bounds__(256) void k_sa(const float* __restrict__ xyz, int xs, int xbs,
        const float* __restrict__ nxyz, const int* __restrict__ nidx,
        const _Float16* __restrict__ feat, int N, int Cf,
        const _Float16* W1, const float* b1,
        const _Float16* W2, const float* b2,
        const _Float16* W3, const float* b3,
        int C3, _Float16* __restrict__ outF, int S) {
    __shared__ __align__(16) _Float16 A0[64 * 168];
    __shared__ __align__(16) _Float16 A1[64 * 136];
    __shared__ __align__(16) _Float16 A2[64 * 136];
    __shared__ int   sidx[64];
    __shared__ float scen[3];
    const int tid = threadIdx.x;
    const int g   = blockIdx.x;
    const int b   = g / S;
    if (tid < 64) sidx[tid] = nidx[(size_t)g * 64 + tid];
    if (tid < 3)  scen[tid] = nxyz[(size_t)g * 3 + tid];
    __syncthreads();
    for (int p = tid; p < 64 * 168; p += 256) {
        int j = p / 168, k = p % 168;
        int id = sidx[j];
        float v;
        if (k < 3)            v = xyz[(size_t)b * xbs + (size_t)id * xs + k] - scen[k];
        else if (k < 3 + Cf)  v = (float)feat[((size_t)b * N + id) * Cf + (k - 3)];
        else                  v = 0.f;
        A0[(size_t)j * 168 + k] = (_Float16)v;
    }
    __syncthreads();
    const int w = tid >> 5;
    {
        v8f acc[4] = {v8f_zero(), v8f_zero(), v8f_zero(), v8f_zero()};
        wmma_block<4>(A0, 168, W1, 160, w, 5, acc);
        store_relu_lds(A1, 136, w, b1, acc, 4);
    }
    __syncthreads();
    {
        v8f acc[4] = {v8f_zero(), v8f_zero(), v8f_zero(), v8f_zero()};
        wmma_block<4>(A1, 136, W2, 128, w, 4, acc);
        store_relu_lds(A2, 136, w, b2, acc, 4);
    }
    __syncthreads();
    const int nt3 = C3 >> 4;
    for (int nt = w; nt < nt3; nt += 8) {
        v8f acc[4] = {v8f_zero(), v8f_zero(), v8f_zero(), v8f_zero()};
        wmma_block<4>(A2, 136, W3, 128, nt, 4, acc);
        const int lane = tid & 31;
        const int col  = lane & 15;
        const int hi   = lane >> 4;
        float m = -3.4e38f;
        for (int mt = 0; mt < 4; ++mt)
#pragma unroll
            for (int v = 0; v < 8; ++v) m = fmaxf(m, acc[mt][v]);
        float o = __shfl_xor(m, 16, 32);
        m = fmaxf(m, o);
        float x = m + b3[nt * 16 + col];
        x = x > 0.f ? x : 0.f;
        if (hi == 0) outF[(size_t)g * C3 + nt * 16 + col] = (_Float16)x;
    }
}

// ---------------- global SA: 259->256->256->512, max over 32 rows -----------
__global__ __launch_bounds__(256) void k_sa2(const float* __restrict__ nxyz1,
        const _Float16* __restrict__ feat2,
        const _Float16* W1, const float* b1,
        const _Float16* W2, const float* b2,
        const _Float16* W3, const float* b3,
        _Float16* __restrict__ fvec) {
    __shared__ __align__(16) _Float16 A0[32 * 296];
    __shared__ __align__(16) _Float16 A1[32 * 264];
    __shared__ __align__(16) _Float16 A2[32 * 264];
    const int tid = threadIdx.x;
    const int b   = blockIdx.x;
    for (int p = tid; p < 32 * 296; p += 256) {
        int j = p / 296, k = p % 296;
        float v;
        if (k < 3)        v = nxyz1[((size_t)b * 32 + j) * 3 + k];
        else if (k < 259) v = (float)feat2[((size_t)b * 32 + j) * 256 + (k - 3)];
        else              v = 0.f;
        A0[(size_t)j * 296 + k] = (_Float16)v;
    }
    __syncthreads();
    const int w = tid >> 5;
    for (int nt = w; nt < 16; nt += 8) {
        v8f acc[2] = {v8f_zero(), v8f_zero()};
        wmma_block<2>(A0, 296, W1, 288, nt, 9, acc);
        store_relu_lds(A1, 264, nt, b1, acc, 2);
    }
    __syncthreads();
    for (int nt = w; nt < 16; nt += 8) {
        v8f acc[2] = {v8f_zero(), v8f_zero()};
        wmma_block<2>(A1, 264, W2, 256, nt, 8, acc);
        store_relu_lds(A2, 264, nt, b2, acc, 2);
    }
    __syncthreads();
    for (int nt = w; nt < 32; nt += 8) {
        v8f acc[2] = {v8f_zero(), v8f_zero()};
        wmma_block<2>(A2, 264, W3, 256, nt, 8, acc);
        const int lane = tid & 31;
        const int col  = lane & 15;
        const int hi   = lane >> 4;
        float m = -3.4e38f;
        for (int mt = 0; mt < 2; ++mt)
#pragma unroll
            for (int v = 0; v < 8; ++v) m = fmaxf(m, acc[mt][v]);
        float o = __shfl_xor(m, 16, 32);
        m = fmaxf(m, o);
        float x = m + b3[nt * 16 + col];
        x = x > 0.f ? x : 0.f;
        if (hi == 0) fvec[(size_t)b * 512 + nt * 16 + col] = (_Float16)x;
    }
}

// ---------------- head hidden GEMM: out = relu(A x W^T + b), f16 ------------
__global__ __launch_bounds__(256) void k_gemm_relu(const _Float16* __restrict__ A, int K,
        const _Float16* __restrict__ Wt, int Nc, const float* __restrict__ bias,
        _Float16* __restrict__ out) {
    __shared__ __align__(16) _Float16 As[32 * 520];
    const int tid = threadIdx.x;
    const int r0  = blockIdx.x * 32;
    const int lda = K + 8;
    for (int p = tid; p < 32 * K; p += 256) {
        int row = p / K, k = p % K;
        As[(size_t)row * lda + k] = A[(size_t)(r0 + row) * K + k];
    }
    __syncthreads();
    const int w = tid >> 5;
    const int KT = K >> 5;
    const int nt3 = Nc >> 4;
    for (int nt = w; nt < nt3; nt += 8) {
        v8f acc[2] = {v8f_zero(), v8f_zero()};
        wmma_block<2>(As, lda, Wt, K, nt, KT, acc);
        const int lane = tid & 31;
        const int col  = nt * 16 + (lane & 15);
        const int hi   = lane >> 4;
        const float bb = bias[col];
        for (int mt = 0; mt < 2; ++mt)
#pragma unroll
            for (int v = 0; v < 8; ++v) {
                float x = acc[mt][v] + bb;
                x = x > 0.f ? x : 0.f;
                out[(size_t)(r0 + mt * 16 + v + 8 * hi) * Nc + col] = (_Float16)x;
            }
    }
}

// ---------------- tiny final projection (no activation), f32 out ------------
__global__ void k_proj(const _Float16* __restrict__ A, const float* __restrict__ Wt,
                       const float* __restrict__ bias, float* __restrict__ out,
                       int M, int N, int K) {
    int i = blockIdx.x * blockDim.x + threadIdx.x;
    if (i >= M * N) return;
    int m = i / N, n = i % N;
    float s = bias[n];
    for (int k = 0; k < K; ++k)
        s += (float)A[(size_t)m * K + k] * Wt[(size_t)n * K + k];
    out[i] = s;
}

// ============================================================================
extern "C" void kernel_launch(void* const* d_in, const int* in_sizes, int n_in,
                              void* d_out, int out_size, void* d_ws, size_t ws_size,
                              hipStream_t stream) {
    (void)in_sizes; (void)n_in; (void)out_size; (void)ws_size;
    const float* pts = (const float*)d_in[0];
    auto F = [&](int i) { return (const float*)d_in[i]; };

    char* base = (char*)d_ws;
    size_t off = 0;
    auto alloc = [&](size_t bytes) -> void* {
        off = (off + 255) & ~(size_t)255;
        void* p = base + off;
        off += bytes;
        return p;
    };
    auto conv = [&](int di, int N, int K, int Kp) -> _Float16* {
        _Float16* dst = (_Float16*)alloc((size_t)N * Kp * 2);
        int total = N * Kp;
        k_convpad<<<(total + 255) / 256, 256, 0, stream>>>(F(di), dst, N, K, Kp);
        return dst;
    };

    // f16 weights (K padded to mult of 32)
    _Float16* w_xyz0 = conv(1, 128, 5, 32);
    _Float16* w_xyz1 = conv(3, 128, 128, 128);
    _Float16* w_mrg  = conv(5, 128, 256, 256);
    _Float16* w_s00  = conv(7, 128, 131, 160);
    _Float16* w_s01  = conv(9, 128, 128, 128);
    _Float16* w_s02  = conv(11, 128, 128, 128);
    _Float16* w_s10  = conv(13, 128, 131, 160);
    _Float16* w_s11  = conv(15, 128, 128, 128);
    _Float16* w_s12  = conv(17, 256, 128, 128);
    _Float16* w_s20  = conv(19, 256, 259, 288);
    _Float16* w_s21  = conv(21, 256, 256, 256);
    _Float16* w_s22  = conv(23, 512, 256, 256);
    _Float16* w_c0   = conv(25, 256, 512, 512);
    _Float16* w_c1   = conv(27, 256, 256, 256);
    _Float16* w_r0   = conv(31, 256, 512, 512);
    _Float16* w_r1   = conv(33, 256, 256, 256);
    _Float16* w_i0   = conv(37, 512, 512, 512);

    const int B = 128, N0 = 512, S0 = 128, S1 = 32;

    _Float16* feat0 = (_Float16*)alloc((size_t)B * N0 * 128 * 2);
    int*      fps0  = (int*)alloc((size_t)B * S0 * 4);
    float*    nx0   = (float*)alloc((size_t)B * S0 * 3 * 4);
    int*      nid0  = (int*)alloc((size_t)B * S0 * 64 * 4);
    _Float16* feat1 = (_Float16*)alloc((size_t)B * S0 * 128 * 2);
    int*      fps1  = (int*)alloc((size_t)B * S1 * 4);
    float*    nx1   = (float*)alloc((size_t)B * S1 * 3 * 4);
    int*      nid1  = (int*)alloc((size_t)B * S1 * 64 * 4);
    _Float16* feat2 = (_Float16*)alloc((size_t)B * S1 * 256 * 2);
    _Float16* fvec  = (_Float16*)alloc((size_t)B * 512 * 2);
    _Float16* ch1   = (_Float16*)alloc((size_t)B * 256 * 2);
    _Float16* ch2   = (_Float16*)alloc((size_t)B * 256 * 2);
    _Float16* rh1   = (_Float16*)alloc((size_t)B * 256 * 2);
    _Float16* rh2   = (_Float16*)alloc((size_t)B * 256 * 2);
    _Float16* ih1   = (_Float16*)alloc((size_t)B * 512 * 2);

    // stage 1: per-point feature MLP
    k_pointmlp<<<(B * N0) / 64, 256, 0, stream>>>(pts, w_xyz0, F(2), w_xyz1, F(4),
                                                  w_mrg, F(6), feat0);

    // SA0: xyz is pts rows (stride 133), N=512 -> S=128, r=0.2, ns=64
    k_fps<<<B, 256, 0, stream>>>(pts, 133, N0 * 133, N0, S0, fps0);
    k_gather_nx<<<(B * S0 + 255) / 256, 256, 0, stream>>>(pts, 133, N0 * 133, fps0, S0,
                                                          nx0, B * S0);
    k_ballq<<<(B * S0 + 255) / 256, 256, 0, stream>>>(pts, 133, N0 * 133, N0, nx0, S0,
                                                      0.04f, nid0, B * S0);
    k_sa<<<B * S0, 256, 0, stream>>>(pts, 133, N0 * 133, nx0, nid0, feat0, N0, 128,
                                     w_s00, F(8), w_s01, F(10), w_s02, F(12),
                                     128, feat1, S0);

    // SA1: xyz = nx0 (stride 3), N=128 -> S=32, r=0.4
    k_fps<<<B, 256, 0, stream>>>(nx0, 3, S0 * 3, S0, S1, fps1);
    k_gather_nx<<<(B * S1 + 255) / 256, 256, 0, stream>>>(nx0, 3, S0 * 3, fps1, S1,
                                                          nx1, B * S1);
    k_ballq<<<(B * S1 + 255) / 256, 256, 0, stream>>>(nx0, 3, S0 * 3, S0, nx1, S1,
                                                      0.16f, nid1, B * S1);
    k_sa<<<B * S1, 256, 0, stream>>>(nx0, 3, S0 * 3, nx1, nid1, feat1, S0, 128,
                                     w_s10, F(14), w_s11, F(16), w_s12, F(18),
                                     256, feat2, S1);

    // SA2: global, 259 -> 256 -> 256 -> 512, max over 32
    k_sa2<<<B, 256, 0, stream>>>(nx1, feat2, w_s20, F(20), w_s21, F(22), w_s22, F(24),
                                 fvec);

    float* out_cls = (float*)d_out;
    float* out_reg = out_cls + 128;
    float* out_iou = out_reg + 128 * 46;

    // cls head
    k_gemm_relu<<<B / 32, 256, 0, stream>>>(fvec, 512, w_c0, 256, F(26), ch1);
    k_gemm_relu<<<B / 32, 256, 0, stream>>>(ch1, 256, w_c1, 256, F(28), ch2);
    k_proj<<<(B * 1 + 255) / 256, 256, 0, stream>>>(ch2, F(29), F(30), out_cls, B, 1, 256);
    // reg head
    k_gemm_relu<<<B / 32, 256, 0, stream>>>(fvec, 512, w_r0, 256, F(32), rh1);
    k_gemm_relu<<<B / 32, 256, 0, stream>>>(rh1, 256, w_r1, 256, F(34), rh2);
    k_proj<<<(B * 46 + 255) / 256, 256, 0, stream>>>(rh2, F(35), F(36), out_reg, B, 46, 256);
    // iou head
    k_gemm_relu<<<B / 32, 256, 0, stream>>>(fvec, 512, w_i0, 512, F(38), ih1);
    k_proj<<<(B * 1 + 255) / 256, 256, 0, stream>>>(ih1, F(39), F(40), out_iou, B, 1, 512);
}
